// NodeModelIn_42056319762636
// MI455X (gfx1250) — matrix operand order repaired and checked by reading, not compiled, not run on hardware
//
#include <hip/hip_runtime.h>
#include <hip/hip_bf16.h>

typedef _Float16 v16h __attribute__((ext_vector_type(16)));
typedef _Float16 v8h  __attribute__((ext_vector_type(8)));
typedef float    v8f  __attribute__((ext_vector_type(8)));

#define N_NODES 50000
#define N_EDGES 800000
#define EOUT    96
#define HID     256
#define NOUT    128
#define K1PAD   320          // 289 padded up to multiple of 32
#define FEAT_LD 328          // LDS row stride (halves), 656B = 41*16B
#define H_LD    264          // LDS row stride (halves), 528B = 33*16B
#define RT      4            // row-tiles (of 16 nodes) per wave: B-fragment reuse x4
#define NTILES  ((N_NODES + 15) / 16)        // 3125

// workspace byte offsets
#define OFF_S    0
#define OFF_MX   (N_NODES * EOUT * 4)                      // 19,200,000
#define OFF_CNT  (2 * N_NODES * EOUT * 4)                  // 38,400,000
#define OFF_W1P  (OFF_CNT + N_NODES * 4)                   // 38,600,000 (32B aligned)
#define OFF_W2P  (OFF_W1P + (K1PAD/32)*(HID/16)*32*16*2)   // +163,840

__global__ void k_init(float* __restrict__ s, unsigned int* __restrict__ mx,
                       float* __restrict__ cnt) {
  int i = blockIdx.x * blockDim.x + threadIdx.x;
  if (i < N_NODES * EOUT) { s[i] = 0.f; mx[i] = 0xFF800000u; /* -inf */ }
  if (i < N_NODES) cnt[i] = 0.f;
}

// Pack W1 (289x256, zero-padded to 320) and W2 (256x128) into f16 B-fragment
// layout: dst = ((chunk*NT + tile)*32 + lane)*16 + h  where for element (k,n):
//   chunk=k>>5, tile=n>>4, lane=(n&15)+((k&16)?16:0), h=k&15
__global__ void k_pack(const float* __restrict__ W1, const float* __restrict__ W2,
                       _Float16* __restrict__ w1p, _Float16* __restrict__ w2p) {
  int i = blockIdx.x * blockDim.x + threadIdx.x;
  if (i < K1PAD * HID) {
    int k = i / HID, n = i % HID;
    float v = (k < 289) ? W1[k * HID + n] : 0.f;
    int c = k >> 5, kk = k & 31, t = n >> 4, nn = n & 15;
    int lane = nn + ((kk & 16) ? 16 : 0);
    w1p[((c * (HID/16) + t) * 32 + lane) * 16 + (kk & 15)] = (_Float16)v;
  } else if (i < K1PAD * HID + HID * NOUT) {
    int j = i - K1PAD * HID;
    int k = j / NOUT, n = j % NOUT;
    float v = W2[k * NOUT + n];
    int c = k >> 5, kk = k & 31, t = n >> 4, nn = n & 15;
    int lane = nn + ((kk & 16) ? 16 : 0);
    w2p[((c * (NOUT/16) + t) * 32 + lane) * 16 + (kk & 15)] = (_Float16)v;
  }
}

__device__ inline void atomicMaxF(float* addr, float v) {
  // classic bit-pun trick: works with -inf init, mixes pos/neg correctly
  if (v >= 0.f) atomicMax((int*)addr, __float_as_int(v));
  else          atomicMin((unsigned int*)addr, __float_as_uint(v));
}

__global__ void k_scatter(const float* __restrict__ ea, const int* __restrict__ col,
                          float* __restrict__ s, float* __restrict__ mx,
                          float* __restrict__ cnt) {
  int gid = blockIdx.x * blockDim.x + threadIdx.x;
  if (gid >= N_EDGES * EOUT) return;
  int e = gid / EOUT;
  int f = gid - e * EOUT;
  int c = col[e];
  float v = ea[gid];
  atomicAdd(&s[c * EOUT + f], v);
  atomicMaxF(&mx[c * EOUT + f], v);
  if (f == 0) atomicAdd(&cnt[c], 1.f);
}

// Build v16h A-fragment: per-lane halves 0..15 = K kb..kb+15
__device__ inline v16h frag_a(const _Float16* p) {
  v8h lo = *(const v8h*)(p);
  v8h hi = *(const v8h*)(p + 8);
  v16h a;
#pragma unroll
  for (int i = 0; i < 8; ++i) { a[i] = lo[i]; a[i + 8] = hi[i]; }
  return a;
}

__global__ __launch_bounds__(32) void k_mlp(
    const float* __restrict__ s, const float* __restrict__ mx,
    const float* __restrict__ cnt,
    const _Float16* __restrict__ w1p, const _Float16* __restrict__ w2p,
    const float* __restrict__ b1, const float* __restrict__ b2,
    const float* __restrict__ u, const int* __restrict__ batch,
    float* __restrict__ out) {
  __shared__ _Float16 featL[RT * 16 * FEAT_LD];   // 41,984 B
  __shared__ _Float16 hL[RT * 16 * H_LD];         // 33,792 B
  const int lane  = threadIdx.x;
  const int node0 = blockIdx.x * (16 * RT);

  // ---- build RT x 16 x 320 f16 feature tile: [sum | masked max | mean | u | 0s]
  for (int idx = lane; idx < RT * 16 * EOUT; idx += 32) {
    int r = idx / EOUT, f = idx - r * EOUT;
    int nd = node0 + r;
    float sv = 0.f, mv = 0.f, mean = 0.f;
    if (nd < N_NODES) {
      sv = s[nd * EOUT + f];
      float c = cnt[nd];
      mv = (c > 0.f) ? mx[nd * EOUT + f] : 0.f;
      mean = sv / fmaxf(c, 1.f);
    }
    featL[r * FEAT_LD + f]            = (_Float16)sv;
    featL[r * FEAT_LD + EOUT + f]     = (_Float16)mv;
    featL[r * FEAT_LD + 2 * EOUT + f] = (_Float16)mean;
  }
  for (int r = lane; r < RT * 16; r += 32) {
    int nd = node0 + r;
    float uv = (nd < N_NODES) ? u[batch[nd]] : 0.f;
    featL[r * FEAT_LD + 288] = (_Float16)uv;
#pragma unroll
    for (int k = 289; k < K1PAD; ++k) featL[r * FEAT_LD + k] = (_Float16)0.f;
  }
  __syncthreads();

  const int n   = lane & 15;          // output column within tile (C/D layout)
  const int mh  = (lane >> 4) * 8;    // row base within tile (C/D layout)
  const int kb  = (lane >> 4) * 16;   // A-fragment K base within chunk
  const int row = lane & 15;          // A-fragment row within tile

  // ---- layer 1: h = relu(feat @ W1 + b1): 16 col-tiles x 10 K-chunks x RT rows
  for (int t = 0; t < HID / 16; ++t) {
    v8f acc[RT] = {};
    const _Float16* wp = w1p + (t * 32 + lane) * 16;
#pragma unroll
    for (int c = 0; c < K1PAD / 32; ++c) {
      v16h b = *(const v16h*)(wp + c * (HID / 16) * 32 * 16);
#pragma unroll
      for (int r = 0; r < RT; ++r) {
        v16h a = frag_a(featL + (r * 16 + row) * FEAT_LD + c * 32 + kb);
        acc[r] = __builtin_amdgcn_wmma_f32_16x16x32_f16(false, a, false, b,
                                                        (short)0, acc[r], false, false);
      }
    }
    float bb = b1[t * 16 + n];
#pragma unroll
    for (int r = 0; r < RT; ++r)
#pragma unroll
      for (int j = 0; j < 8; ++j) {
        float hv = fmaxf(acc[r][j] + bb, 0.f);
        hL[(r * 16 + mh + j) * H_LD + t * 16 + n] = (_Float16)hv;
      }
  }
  __syncthreads();

  // ---- layer 2: out = h @ W2 + b2: 8 col-tiles x 8 K-chunks x RT rows
  for (int t = 0; t < NOUT / 16; ++t) {
    v8f acc[RT] = {};
    const _Float16* wp = w2p + (t * 32 + lane) * 16;
#pragma unroll
    for (int c = 0; c < HID / 32; ++c) {
      v16h b = *(const v16h*)(wp + c * (NOUT / 16) * 32 * 16);
#pragma unroll
      for (int r = 0; r < RT; ++r) {
        v16h a = frag_a(hL + (r * 16 + row) * H_LD + c * 32 + kb);
        acc[r] = __builtin_amdgcn_wmma_f32_16x16x32_f16(false, a, false, b,
                                                        (short)0, acc[r], false, false);
      }
    }
    float bb = b2[t * 16 + n];
#pragma unroll
    for (int r = 0; r < RT; ++r) {
#pragma unroll
      for (int j = 0; j < 8; ++j) {
        int nd = node0 + r * 16 + mh + j;
        if (nd < N_NODES) out[nd * NOUT + t * 16 + n] = acc[r][j] + bb;
      }
    }
  }
}

extern "C" void kernel_launch(void* const* d_in, const int* in_sizes, int n_in,
                              void* d_out, int out_size, void* d_ws, size_t ws_size,
                              hipStream_t stream) {
  (void)in_sizes; (void)n_in; (void)out_size; (void)ws_size;
  const float* x          = (const float*)d_in[0]; (void)x; // unused by reference math
  const int*   edge_index = (const int*)d_in[1];
  const float* edge_attr  = (const float*)d_in[2];
  const float* u          = (const float*)d_in[3];
  const int*   batch      = (const int*)d_in[4];
  const float* W1         = (const float*)d_in[5];
  const float* b1         = (const float*)d_in[6];
  const float* W2         = (const float*)d_in[7];
  const float* b2         = (const float*)d_in[8];
  float* out = (float*)d_out;

  char* ws = (char*)d_ws;
  float*    d_s   = (float*)(ws + OFF_S);
  float*    d_mx  = (float*)(ws + OFF_MX);
  float*    d_cnt = (float*)(ws + OFF_CNT);
  _Float16* d_w1p = (_Float16*)(ws + OFF_W1P);
  _Float16* d_w2p = (_Float16*)(ws + OFF_W2P);

  const int* col = edge_index + N_EDGES;  // edge_index[1]

  k_init<<<(N_NODES * EOUT + 255) / 256, 256, 0, stream>>>(
      d_s, (unsigned int*)d_mx, d_cnt);
  k_pack<<<(K1PAD * HID + HID * NOUT + 255) / 256, 256, 0, stream>>>(
      W1, W2, d_w1p, d_w2p);
  k_scatter<<<(N_EDGES * EOUT + 255) / 256, 256, 0, stream>>>(
      edge_attr, col, d_s, d_mx, d_cnt);
  k_mlp<<<(NTILES + RT - 1) / RT, 32, 0, stream>>>(
      d_s, d_mx, d_cnt, d_w1p, d_w2p, b1, b2, u, batch, out);
}
